// SpectralConv2d_11862699671592
// MI455X (gfx1250) — matrix operand order repaired and checked by reading, not compiled, not run on hardware
//
#include <hip/hip_runtime.h>
#include <hip/hip_bf16.h>

// ---------------------------------------------------------------------------
// SpectralConv2d on MI455X (gfx1250).
//   B=16, C_in=C_out=32, H=W=256, k=16.
//   ws layout (floats): Sre[8192] | Sim[8192] | BLKre[131072] | BLKim[131072]
// ---------------------------------------------------------------------------

#define PI_F 3.14159265358979323846f

typedef __attribute__((ext_vector_type(16))) _Float16 v16h;
typedef __attribute__((ext_vector_type(8)))  float    v8f;

// ---------------------------------------------------------------------------
// Kernel 1: stream x (128 MiB), reduce rows with G(h), 16-pt column DFT -> S.
// One workgroup per (b, c_in) slice (512 blocks x 256 threads).
// ---------------------------------------------------------------------------
__global__ __launch_bounds__(256) void k_spectral_S(
    const float* __restrict__ x, float* __restrict__ Sre, float* __restrict__ Sim) {
  __shared__ float Gre[256], Gim[256], Xre[256], Xim[256];
  const int tid = threadIdx.x;
  const int bc  = blockIdx.x;  // b*32 + c_in

  // G(h) = sum_{u=0..15} e^{-2*pi*i*u*h/256} for h = tid
  {
    float ph = -2.0f * PI_F * (float)tid * (1.0f / 256.0f);
    float s1, c1;
    __sincosf(ph, &s1, &c1);
    float cu = 1.0f, su = 0.0f, gr = 0.0f, gi = 0.0f;
#pragma unroll
    for (int u = 0; u < 16; ++u) {
      gr += cu; gi += su;
      float nc = cu * c1 - su * s1;
      su = cu * s1 + su * c1;
      cu = nc;
    }
    Gre[tid] = gr; Gim[tid] = gi;
  }
  __syncthreads();

  // X[w] = sum_h G(h) * x[h,w]   (thread tid owns column w = tid; coalesced rows)
  const float* xp = x + (size_t)bc * 65536 + tid;
  float ar = 0.0f, ai = 0.0f;
#pragma unroll 4
  for (int h = 0; h < 256; ++h) {
    if ((h & 3) == 0)  // global_prefetch_b8, ~16 KB ahead (speculative; OOB dropped)
      __builtin_prefetch(xp + (size_t)(h + 16) * 256, 0, 0);
    float v = xp[(size_t)h * 256];
    ar = fmaf(Gre[h], v, ar);
    ai = fmaf(Gim[h], v, ai);
  }
  Xre[tid] = ar; Xim[tid] = ai;
  __syncthreads();

  // S[r] = (1/256) * sum_w X[w] * e^{-2*pi*i*r*w/256}, r = 0..15
  if (tid < 16) {
    const int r = tid;
    float ph = -2.0f * PI_F * (float)r * (1.0f / 256.0f);
    float s1, c1;
    __sincosf(ph, &s1, &c1);
    float cw = 1.0f, sw = 0.0f, sr = 0.0f, si = 0.0f;
    for (int w = 0; w < 256; ++w) {
      float xr = Xre[w], xi = Xim[w];
      sr = fmaf(xr, cw, fmaf(-xi, sw, sr));
      si = fmaf(xr, sw, fmaf( xi, cw, si));
      float nc = cw * c1 - sw * s1;
      sw = cw * s1 + sw * c1;
      cw = nc;
    }
    Sre[bc * 16 + r] = sr * (1.0f / 256.0f);
    Sim[bc * 16 + r] = si * (1.0f / 256.0f);
  }
}

// ---------------------------------------------------------------------------
// Kernel 2: channel mix.  blk[b,o,r,c] = sum_i S[b,i,r] * (wr + i*wi)[i,o,r,c]
// 512 blocks (b*32+o) x 256 threads (r*16+c).
// ---------------------------------------------------------------------------
__global__ __launch_bounds__(256) void k_channel_mix(
    const float* __restrict__ Sre, const float* __restrict__ Sim,
    const float* __restrict__ wr, const float* __restrict__ wi,
    float* __restrict__ BLKre, float* __restrict__ BLKim) {
  const int tid = threadIdx.x;      // r*16 + c
  const int bo  = blockIdx.x;       // b*32 + o
  const int b   = bo >> 5;
  const int o   = bo & 31;
  const int r   = tid >> 4;

  float br = 0.0f, bi = 0.0f;
#pragma unroll 4
  for (int i = 0; i < 32; ++i) {
    float sr = Sre[(b * 32 + i) * 16 + r];
    float si = Sim[(b * 32 + i) * 16 + r];
    float wrv = wr[(i * 32 + o) * 256 + tid];
    float wiv = wi[(i * 32 + o) * 256 + tid];
    br = fmaf(sr, wrv, fmaf(-si, wiv, br));
    bi = fmaf(sr, wiv, fmaf( si, wrv, bi));
  }
  BLKre[bo * 256 + tid] = br;
  BLKim[bo * 256 + tid] = bi;
}

// ---------------------------------------------------------------------------
// Kernel 3: inverse transform as WMMA GEMM.
// Per (b,o): E[h,v] = (1/16) sum_u blk[u,v] e^{2*pi*i*u*h/256}  (stage A, VALU)
// then out(256x256) = [Ere|Eim](256x32) @ [c_v cos/16 ; -c_v sin/16](32x256)
// via v_wmma_f32_16x16x32_f16 (K = 32 exactly).  512 blocks x 256 threads.
// Each wave pins 2 column-tile B fragments (trig) in registers and streams
// all 16 row tiles: 2 WMMAs per A-fragment load.
// ---------------------------------------------------------------------------
#define AB_STRIDE 34   // halves per row; 34*2B = 68B => conflict-free-ish banks

__global__ __launch_bounds__(256) void k_inverse_wmma(
    const float* __restrict__ BLKre, const float* __restrict__ BLKim,
    float* __restrict__ out) {
  __shared__ _Float16 Amat[256 * AB_STRIDE];   // [h][k]  k<16: Ere, k>=16: Eim
  __shared__ _Float16 BmatT[256 * AB_STRIDE];  // [w][k]  k<16: c*cos, k>=16: -c*sin
  __shared__ float    bre[256], bim[256];

  const int tid = threadIdx.x;
  const int bo  = blockIdx.x;  // b*32 + o

  bre[tid] = BLKre[bo * 256 + tid];
  bim[tid] = BLKim[bo * 256 + tid];
  __syncthreads();

  // ---- Stage A: per-thread row h = tid of E, stored as f16 A-operand ----
  {
    const int h = tid;
    float er[16], ei[16];
#pragma unroll
    for (int v = 0; v < 16; ++v) { er[v] = 0.0f; ei[v] = 0.0f; }
    float ph = 2.0f * PI_F * (float)h * (1.0f / 256.0f);
    float s1, c1;
    __sincosf(ph, &s1, &c1);
    float cu = 1.0f, su = 0.0f;
#pragma unroll
    for (int u = 0; u < 16; ++u) {
#pragma unroll
      for (int v = 0; v < 16; ++v) {
        float rr = bre[u * 16 + v], ii = bim[u * 16 + v];
        er[v] = fmaf(rr, cu, fmaf(-ii, su, er[v]));
        ei[v] = fmaf(rr, su, fmaf( ii, cu, ei[v]));
      }
      float nc = cu * c1 - su * s1;
      su = cu * s1 + su * c1;
      cu = nc;
    }
#pragma unroll
    for (int v = 0; v < 16; ++v) {
      Amat[h * AB_STRIDE + v]      = (_Float16)(er[v] * (1.0f / 16.0f));
      Amat[h * AB_STRIDE + 16 + v] = (_Float16)(ei[v] * (1.0f / 16.0f));
    }
  }

  // ---- Stage B table: column w = tid of the trig B-operand (K-contiguous) ----
  {
    const int w = tid;
#pragma unroll
    for (int v = 0; v < 16; ++v) {
      float ph = 2.0f * PI_F * (float)(v * w) * (1.0f / 256.0f);
      float s, c;
      __sincosf(ph, &s, &c);
      float f = ((v == 0) ? 1.0f : 2.0f) * (1.0f / 16.0f);
      BmatT[w * AB_STRIDE + v]      = (_Float16)(f * c);
      BmatT[w * AB_STRIDE + 16 + v] = (_Float16)((v == 0) ? 0.0f : (-f * s));
    }
  }
  __syncthreads();

  // ---- Stage C: 16x16 output tiles via WMMA f32 <- f16xf16, K=32 ----
  const int wv   = tid >> 5;         // wave 0..7
  const int lane = tid & 31;
  const int half = lane >> 4;        // 0: lanes 0-15, 1: lanes 16-31
  const int ln   = lane & 15;
  const int ko   = half * 8;         // A K-offset per lane group

  // Wave wv owns column tiles nt0, nt0+1: B fragments pinned in registers.
  const int nt0 = wv * 2;
  v16h b0, b1;
  {
    const _Float16* br0 = &BmatT[((nt0    ) * 16 + ln) * AB_STRIDE + half * 8];
    const _Float16* br1 = &BmatT[((nt0 + 1) * 16 + ln) * AB_STRIDE + half * 8];
#pragma unroll
    for (int j = 0; j < 8; ++j) { b0[j] = br0[j]; b1[j] = br1[j]; b0[8 + j] = br0[16 + j]; b1[8 + j] = br1[16 + j]; }
  }

  float* outp = out + (size_t)bo * 65536;
  const int col0 = nt0 * 16 + ln;

  for (int mt = 0; mt < 16; ++mt) {
    // A fragment: lane holds row m = mt*16+ln (halves: K ko..ko+7, 16+ko..16+ko+7)
    v16h a;
    const _Float16* arow = &Amat[(mt * 16 + ln) * AB_STRIDE];
#pragma unroll
    for (int j = 0; j < 8; ++j) {
      a[j]     = arow[ko + j];
      a[8 + j] = arow[16 + ko + j];
    }

    v8f acc0 = {}, acc1 = {};
    acc0 = __builtin_amdgcn_wmma_f32_16x16x32_f16(
        false, a, false, b0, (short)0, acc0, false, false);
    acc1 = __builtin_amdgcn_wmma_f32_16x16x32_f16(
        false, a, false, b1, (short)0, acc1, false, false);

    // D layout: lane (ln) = col, rows m = mt*16 + half*8 + j
    const int rowb = mt * 16 + half * 8;
    float* p = outp + (size_t)rowb * 256 + col0;
#pragma unroll
    for (int j = 0; j < 8; ++j) {
      p[(size_t)j * 256]      = acc0[j];
      p[(size_t)j * 256 + 16] = acc1[j];
    }
  }
}

// ---------------------------------------------------------------------------
extern "C" void kernel_launch(void* const* d_in, const int* in_sizes, int n_in,
                              void* d_out, int out_size, void* d_ws, size_t ws_size,
                              hipStream_t stream) {
  (void)in_sizes; (void)n_in; (void)out_size; (void)ws_size;
  const float* x  = (const float*)d_in[0];   // (16,32,256,256)
  const float* wr = (const float*)d_in[1];   // (32,32,16,16)
  const float* wi = (const float*)d_in[2];   // (32,32,16,16)
  float* out = (float*)d_out;                // (16,32,256,256)

  float* Sre   = (float*)d_ws;               //  8192
  float* Sim   = Sre + 8192;                 //  8192
  float* BLKre = Sim + 8192;                 // 131072
  float* BLKim = BLKre + 131072;             // 131072

  k_spectral_S  <<<512, 256, 0, stream>>>(x, Sre, Sim);
  k_channel_mix <<<512, 256, 0, stream>>>(Sre, Sim, wr, wi, BLKre, BLKim);
  k_inverse_wmma<<<512, 256, 0, stream>>>(BLKre, BLKim, out);
}
